// BiDirectionalMinGRU_85160611545252
// MI455X (gfx1250) — matrix operand order repaired and checked by reading, not compiled, not run on hardware
//
#include <hip/hip_runtime.h>
#include <stdint.h>

// ---------------- problem constants ----------------
constexpr int Bc = 256;
constexpr int Lc = 2048;
constexpr int Hc = 64;
constexpr int TEc = 64;
constexpr long NTOK = (long)Bc * Lc;          // 524288
constexpr long NTILE = NTOK / 16;             // 32768
constexpr int NW = 8;                          // waves per block (256 threads)

// ---------------- WMMA helpers ----------------
typedef _Float16 h8  __attribute__((ext_vector_type(8)));
typedef _Float16 h16 __attribute__((ext_vector_type(16)));
typedef float    f8  __attribute__((ext_vector_type(8)));

__device__ __forceinline__ h16 cat8(h8 lo, h8 hi) {
  return __builtin_shufflevector(lo, hi, 0,1,2,3,4,5,6,7,8,9,10,11,12,13,14,15);
}
// A-fragment (16x32 f16, row-major source, per-lane row m): VGPR0-3 = K h*8..h*8+7,
// VGPR4-7 = K 16+h*8..16+h*8+7
__device__ __forceinline__ h16 ldA(const _Float16* row, int kbase, int halfid) {
  h8 lo = *(const h8*)(row + kbase + halfid * 8);
  h8 hi = *(const h8*)(row + kbase + 16 + halfid * 8);
  return cat8(lo, hi);
}
// B-fragment (32x16 f16). Weights stored [N][K]; lane holds column n = lane&15,
// K h*8..h*8+7 then 16+h*8..16+h*8+7 (matches A).
__device__ __forceinline__ h16 ldB(const _Float16* wrow, int kbase, int halfid) {
  h8 lo = *(const h8*)(wrow + kbase + halfid * 8);
  h8 hi = *(const h8*)(wrow + kbase + 16 + halfid * 8);
  return cat8(lo, hi);
}
__device__ __forceinline__ f8 wmma16(h16 a, h16 b, f8 c) {
  return __builtin_amdgcn_wmma_f32_16x16x32_f16(false, a, false, b, (short)0, c,
                                                false, false);
}
__device__ __forceinline__ uint32_t pack_h2(float a, float b) {
  _Float16 ah = (_Float16)a, bh = (_Float16)b;
  return (uint32_t)__builtin_bit_cast(unsigned short, ah) |
         ((uint32_t)__builtin_bit_cast(unsigned short, bh) << 16);
}
__device__ __forceinline__ float lo_h(uint32_t c) {
  unsigned short s = (unsigned short)(c & 0xffffu);
  return (float)__builtin_bit_cast(_Float16, s);
}
__device__ __forceinline__ float hi_h(uint32_t c) {
  unsigned short s = (unsigned short)(c >> 16);
  return (float)__builtin_bit_cast(_Float16, s);
}

// ---------------- kernel 1: token-parallel GEMMs -> scan coefficients ----------
// LDS layout (half-element offsets unless noted)
constexpr int OFF_TEW2  = 0;                       // 64*64
constexpr int OFF_FPROJ = 4096;                    // 64*96 (zero-padded K 67..95)
constexpr int OFF_BPROJ = 10240;                   // 64*96
constexpr int OFF_FWZ   = 16384;                   // 64*64
constexpr int OFF_FWH   = 20480;
constexpr int OFF_BWZ   = 24576;
constexpr int OFF_BWH   = 28672;
constexpr int OFF_BIAS_B = 65536;                  // bytes: 7*64 floats
constexpr int OFF_XCAT_B = 67328;                  // bytes: NW*16*96 halves
constexpr int OFF_INP_B  = OFF_XCAT_B + NW * 16 * 96 * 2;   // 91904
constexpr int LDS1_BYTES = OFF_INP_B + NW * 16 * 64 * 2;    // 108288

__global__ void stage1_kernel(const float* __restrict__ xG,
                              const float* __restrict__ tG,
                              const float* __restrict__ mtok,
                              const float* __restrict__ te_w1,
                              const float* __restrict__ te_b1,
                              const float* __restrict__ te_w2,
                              const float* __restrict__ te_b2,
                              const float* __restrict__ fproj_w,
                              const float* __restrict__ fproj_b,
                              const float* __restrict__ bproj_w,
                              const float* __restrict__ bproj_b,
                              const float* __restrict__ fwz, const float* __restrict__ fbz,
                              const float* __restrict__ fwh, const float* __restrict__ fbh,
                              const float* __restrict__ bwz, const float* __restrict__ bbz,
                              const float* __restrict__ bwh, const float* __restrict__ bbh,
                              uint32_t* __restrict__ coefF,
                              uint32_t* __restrict__ coefB,
                              _Float16* __restrict__ tenG) {
  extern __shared__ char smem[];
  _Float16* sh   = (_Float16*)smem;
  float*    sbia = (float*)(smem + OFF_BIAS_B);

  const int tid = threadIdx.x;
  // cooperative weight convert f32 -> f16 LDS
  for (int i = tid; i < 64 * 64; i += 256) sh[OFF_TEW2 + i] = (_Float16)te_w2[i];
  for (int i = tid; i < 64 * 96; i += 256) {
    int o = i / 96, k = i - o * 96;
    sh[OFF_FPROJ + i] = (_Float16)(k < 67 ? fproj_w[o * 67 + k] : 0.f);
    sh[OFF_BPROJ + i] = (_Float16)(k < 67 ? bproj_w[o * 67 + k] : 0.f);
  }
  for (int i = tid; i < 64 * 64; i += 256) {
    sh[OFF_FWZ + i] = (_Float16)fwz[i];
    sh[OFF_FWH + i] = (_Float16)fwh[i];
    sh[OFF_BWZ + i] = (_Float16)bwz[i];
    sh[OFF_BWH + i] = (_Float16)bwh[i];
  }
  for (int i = tid; i < 64; i += 256) {
    sbia[i]       = te_b2[i];
    sbia[64 + i]  = fproj_b[i];
    sbia[128 + i] = bproj_b[i];
    sbia[192 + i] = fbz[i];
    sbia[256 + i] = fbh[i];
    sbia[320 + i] = bbz[i];
    sbia[384 + i] = bbh[i];
  }
  __syncthreads();

  const int wave = tid >> 5, lane = tid & 31;
  const int m = lane & 15, hv = lane >> 4;
  const long tile = (long)blockIdx.x * NW + wave;
  const long tok0 = tile * 16;

  _Float16* xc = (_Float16*)(smem + OFF_XCAT_B) + wave * 16 * 96;
  _Float16* ip = (_Float16*)(smem + OFF_INP_B) + wave * 16 * 64;

  // ---- te layer1 A-fragments (16 tokens x 64, built analytically) ----
  float tval = tG[tok0 + m];
  h16 af[2];
#pragma unroll
  for (int s = 0; s < 2; ++s) {
#pragma unroll
    for (int j = 0; j < 8; ++j) {
      int k1 = s * 32 + hv * 8 + j;
      int k2 = s * 32 + 16 + hv * 8 + j;
      af[s][j]     = (_Float16)fmaxf(te_w1[k1] * tval + te_b1[k1], 0.f);
      af[s][8 + j] = (_Float16)fmaxf(te_w1[k2] * tval + te_b1[k2], 0.f);
    }
  }
  // ---- t_enc = te1 @ te_w2^T + b; write into xcat cols 3..66 and scratch ----
#pragma unroll
  for (int j = 0; j < 4; ++j) {
    int n = j * 16 + m;
    f8 acc = {};
#pragma unroll
    for (int s = 0; s < 2; ++s)
      acc = wmma16(af[s], ldB(sh + OFF_TEW2 + n * 64, s * 32, hv), acc);
    float bias = sbia[n];
#pragma unroll
    for (int v = 0; v < 8; ++v) {
      int r = v + 8 * hv;
      float val = acc[v] + bias;
      xc[r * 96 + 3 + n] = (_Float16)val;
      tenG[(tok0 + r) * 64 + n] = (_Float16)val;
    }
  }
  // ---- x3 (mask token) + zero pad ----
  for (int idx = lane; idx < 48; idx += 32) {
    int r = idx / 3, c = idx - r * 3;
    const float* xr = xG + (tok0 + r) * 3;
    float mv = xr[2];
    float val = (c == 2) ? mv : ((mv == 0.f) ? mtok[c] : xr[c]);
    xc[r * 96 + c] = (_Float16)val;
  }
  for (int idx = lane; idx < 16 * 29; idx += 32) {
    int r = idx / 29, c = 67 + (idx - r * 29);
    xc[r * 96 + c] = (_Float16)0.f;
  }
  __syncthreads();

  // A-fragments of x_cat (K = 96 -> 3 steps)
  h16 ax[3];
#pragma unroll
  for (int s = 0; s < 3; ++s) ax[s] = ldA(xc + m * 96, s * 32, hv);

#pragma unroll
  for (int dir = 0; dir < 2; ++dir) {
    const int projOff = dir ? OFF_BPROJ : OFF_FPROJ;
    const int wzOff   = dir ? OFF_BWZ : OFF_FWZ;
    const int whOff   = dir ? OFF_BWH : OFF_FWH;
    const int pbB = dir ? 128 : 64, zbB = dir ? 320 : 192, hbB = dir ? 384 : 256;
    uint32_t* coef = dir ? coefB : coefF;

    // inp = x_cat @ proj^T + b  -> LDS (f16)
#pragma unroll
    for (int j = 0; j < 4; ++j) {
      int n = j * 16 + m;
      f8 acc = {};
#pragma unroll
      for (int s = 0; s < 3; ++s)
        acc = wmma16(ax[s], ldB(sh + projOff + n * 96, s * 32, hv), acc);
      float bias = sbia[pbB + n];
#pragma unroll
      for (int v = 0; v < 8; ++v)
        ip[(v + 8 * hv) * 64 + n] = (_Float16)(acc[v] + bias);
    }
    __syncthreads();

    h16 ai[2];
#pragma unroll
    for (int s = 0; s < 2; ++s) ai[s] = ldA(ip + m * 64, s * 32, hv);

#pragma unroll
    for (int j = 0; j < 4; ++j) {
      int n = j * 16 + m;
      f8 az = {}, ah = {};
#pragma unroll
      for (int s = 0; s < 2; ++s) {
        az = wmma16(ai[s], ldB(sh + wzOff + n * 64, s * 32, hv), az);
        ah = wmma16(ai[s], ldB(sh + whOff + n * 64, s * 32, hv), ah);
      }
      float bz = sbia[zbB + n], bh = sbia[hbB + n];
#pragma unroll
      for (int v = 0; v < 8; ++v) {
        float z  = 1.f / (1.f + __expf(-(az[v] + bz)));
        float ht = tanhf(ah[v] + bh);
        long tok = tok0 + v + 8 * hv;
        coef[tok * 64 + n] = pack_h2(1.f - z, z * ht);
      }
    }
    __syncthreads();   // before overwriting ip with the other direction
  }
}

// ---------------- kernel 2: parallel elementwise scan ----------------
__global__ void scan_kernel(const uint32_t* __restrict__ coefF,
                            const uint32_t* __restrict__ coefB,
                            _Float16* __restrict__ hfG,
                            _Float16* __restrict__ hbG) {
  int tid = blockIdx.x * blockDim.x + threadIdx.x;  // 0..32767
  int dir = tid >> 14;
  int r = tid & 16383;
  long base = (long)(r >> 6) * (long)(Lc * Hc) + (r & 63);
  if (dir == 0) {
    float h = 0.f;
    for (int l = 0; l < Lc; ++l) {
      long idx = base + (long)l * Hc;
      __builtin_prefetch(coefF + idx + 8 * Hc, 0, 1);
      uint32_t c = coefF[idx];
      hfG[idx] = (_Float16)h;                 // pre-update state
      h = lo_h(c) * h + hi_h(c);
    }
  } else {
    float h = 0.f;
    for (int l = Lc - 1; l >= 0; --l) {
      long idx = base + (long)l * Hc;
      __builtin_prefetch(coefB + idx - 8 * Hc, 0, 1);
      uint32_t c = coefB[idx];
      hbG[idx] = (_Float16)h;                 // pre-update state (reverse)
      h = lo_h(c) * h + hi_h(c);
    }
  }
}

// ---------------- kernel 3: gated head ----------------
constexpr int W1_B   = 0;                      // 128*192 halves = 49152 B
constexpr int W2_B   = 49152;                  // 128 f32
constexpr int B1_B   = 49664;                  // 128 f32
constexpr int AT_B   = 50176;                  // NW * 16*192 halves
constexpr int HID_B  = AT_B + NW * 16 * 192 * 2;        // 99328
constexpr int LDS3_BYTES = HID_B + NW * 16 * 128 * 2;   // 132096

__global__ void head_kernel(const float* __restrict__ xG,
                            const _Float16* __restrict__ hfG,
                            const _Float16* __restrict__ hbG,
                            const _Float16* __restrict__ tenG,
                            const float* __restrict__ gh_w1,
                            const float* __restrict__ gh_b1,
                            const float* __restrict__ gh_w2,
                            const float* __restrict__ gh_b2,
                            float* __restrict__ outG) {
  extern __shared__ char smem[];
  _Float16* w1h = (_Float16*)(smem + W1_B);
  float*    w2f = (float*)(smem + W2_B);
  float*    b1f = (float*)(smem + B1_B);

  const int tid = threadIdx.x;
  for (int i = tid; i < 128 * 192; i += 256) w1h[i] = (_Float16)gh_w1[i];
  for (int i = tid; i < 128; i += 256) { w2f[i] = gh_w2[i]; b1f[i] = gh_b1[i]; }
  __syncthreads();

  const int wave = tid >> 5, lane = tid & 31;
  const int m = lane & 15, hv = lane >> 4;
  const long tile = (long)blockIdx.x * NW + wave;
  const long tok0 = tile * 16;
  const long bb = tok0 / Lc;

  _Float16* At  = (_Float16*)(smem + AT_B) + wave * 16 * 192;
  _Float16* hid = (_Float16*)(smem + HID_B) + wave * 16 * 128;

  // Build [h_fwd_apply | h_bwd_apply | t_enc] 16x192 tile, 16B vector copies
#pragma unroll
  for (int it = 0; it < 4; ++it) {
    int chunk = it * 32 + lane;   // 0..127
    int r = chunk >> 3, g = chunk & 7;
    long tok = tok0 + r;
    bool msk = (xG[tok * 3 + 2] == 0.f);
    long tf = msk ? (bb * Lc + (Lc - 1)) : tok;  // h_fwd final = state at l=L-1
    long tb = msk ? (bb * Lc) : tok;             // h_bwd final = state at l=0
    ((uint4*)(At + r * 192))[g]       = ((const uint4*)(hfG + tf * 64))[g];
    ((uint4*)(At + r * 192 + 64))[g]  = ((const uint4*)(hbG + tb * 64))[g];
    ((uint4*)(At + r * 192 + 128))[g] = ((const uint4*)(tenG + tok * 64))[g];
  }
  __syncthreads();

  h16 aA[6];
#pragma unroll
  for (int s = 0; s < 6; ++s) aA[s] = ldA(At + m * 192, s * 32, hv);

#pragma unroll
  for (int j = 0; j < 8; ++j) {
    int n = j * 16 + m;
    f8 acc = {};
#pragma unroll
    for (int s = 0; s < 6; ++s)
      acc = wmma16(aA[s], ldB(w1h + n * 192, s * 32, hv), acc);
    float bias = b1f[n];
#pragma unroll
    for (int v = 0; v < 8; ++v)
      hid[(v + 8 * hv) * 128 + n] = (_Float16)fmaxf(acc[v] + bias, 0.f);
  }
  __syncthreads();

  if (hv == 0) {
    float s = gh_b2[0];
    const _Float16* hr = hid + m * 128;
#pragma unroll 8
    for (int k = 0; k < 128; ++k) s += (float)hr[k] * w2f[k];
    outG[tok0 + m] = s;
  }
}

// ---------------- launcher ----------------
extern "C" void kernel_launch(void* const* d_in, const int* in_sizes, int n_in,
                              void* d_out, int out_size, void* d_ws, size_t ws_size,
                              hipStream_t stream) {
  const float* x      = (const float*)d_in[0];
  const float* t      = (const float*)d_in[1];
  const float* mtok   = (const float*)d_in[2];
  const float* te_w1  = (const float*)d_in[3];
  const float* te_b1  = (const float*)d_in[4];
  const float* te_w2  = (const float*)d_in[5];
  const float* te_b2  = (const float*)d_in[6];
  const float* fpw    = (const float*)d_in[7];
  const float* fpb    = (const float*)d_in[8];
  const float* bpw    = (const float*)d_in[9];
  const float* bpb    = (const float*)d_in[10];
  const float* fwz    = (const float*)d_in[11];
  const float* fbz    = (const float*)d_in[12];
  const float* fwh    = (const float*)d_in[13];
  const float* fbh    = (const float*)d_in[14];
  const float* bwz    = (const float*)d_in[15];
  const float* bbz    = (const float*)d_in[16];
  const float* bwh    = (const float*)d_in[17];
  const float* bbh    = (const float*)d_in[18];
  const float* gh_w1  = (const float*)d_in[19];
  const float* gh_b1  = (const float*)d_in[20];
  const float* gh_w2  = (const float*)d_in[21];
  const float* gh_b2  = (const float*)d_in[22];

  const size_t COEF_BYTES = (size_t)NTOK * Hc * 4;       // 128 MB each
  const size_t TEN_BYTES  = (size_t)NTOK * TEc * 2;      // 64 MB
  const size_t HS_BYTES   = (size_t)NTOK * Hc * 2;       // 64 MB each
  char* ws = (char*)d_ws;
  uint32_t* coefF = (uint32_t*)ws;
  uint32_t* coefB = (uint32_t*)(ws + COEF_BYTES);
  _Float16* ten   = (_Float16*)(ws + 2 * COEF_BYTES);
  _Float16* hf    = (_Float16*)(ws + 2 * COEF_BYTES + TEN_BYTES);
  _Float16* hb    = (_Float16*)(ws + 2 * COEF_BYTES + TEN_BYTES + HS_BYTES);

  dim3 blk(256);
  stage1_kernel<<<dim3(NTILE / NW), blk, LDS1_BYTES, stream>>>(
      x, t, mtok, te_w1, te_b1, te_w2, te_b2, fpw, fpb, bpw, bpb,
      fwz, fbz, fwh, fbh, bwz, bbz, bwh, bbh, coefF, coefB, ten);
  scan_kernel<<<dim3(2 * Bc * Hc / 256), blk, 0, stream>>>(coefF, coefB, hf, hb);
  head_kernel<<<dim3(NTILE / NW), blk, LDS3_BYTES, stream>>>(
      x, hf, hb, ten, gh_w1, gh_b1, gh_w2, gh_b2, (float*)d_out);
}
